// StructureFeatureEncoder_80788334838053
// MI455X (gfx1250) — matrix operand order repaired and checked by reading, not compiled, not run on hardware
//
#include <hip/hip_runtime.h>

typedef __attribute__((ext_vector_type(16))) __bf16 v16bf;
typedef __attribute__((ext_vector_type(16))) unsigned short v16u;
typedef __attribute__((ext_vector_type(8)))  unsigned short v8u;
typedef __attribute__((ext_vector_type(8)))  float v8f;

#define EPSV 1e-5f

__device__ __forceinline__ unsigned short f2bf(float f) {
  union { float f; unsigned int u; } c; c.f = f;
  unsigned int u = c.u + 0x7FFFu + ((c.u >> 16) & 1u);
  return (unsigned short)(u >> 16);
}

__device__ __forceinline__ unsigned int pack2bf(float a, float b) {
  return (unsigned int)f2bf(a) | ((unsigned int)f2bf(b) << 16);
}

__device__ __forceinline__ v8f zero8() {
  v8f z;
#pragma unroll
  for (int i = 0; i < 8; ++i) z[i] = 0.f;
  return z;
}

// A-fragment (16x32 bf16) from LDS: two 16B vector loads (ds_load_b128) + concat.
// Lane layout per CDNA5 ISA 7.12.2: row = lane&15; lanes16-31 K-offset +8;
// VGPR0-3 hold K {0..7}+off, VGPR4-7 hold K {16..23}+off.
__device__ __forceinline__ v16bf load_a_lds(const unsigned short* base, int ld, int kbase) {
  const int lane = threadIdx.x & 31;
  const unsigned short* p = base + (size_t)(lane & 15) * ld + kbase + ((lane >> 4) << 3);
  v8u lo = *(const v8u*)p;          // 16B aligned (ld, kbase multiples keep alignment)
  v8u hi = *(const v8u*)(p + 16);
  v16u t = __builtin_shufflevector(lo, hi, 0,1,2,3,4,5,6,7,8,9,10,11,12,13,14,15);
  return __builtin_bit_cast(v16bf, t);
}

// B-fragment from pre-packed fragment-major buffer: each lane's 16 bf16 are
// contiguous 32 bytes -> two global_load_b128 instead of 16 global_load_u16.
__device__ __forceinline__ v16bf load_b_pk(const unsigned short* __restrict__ packed,
                                           int kt, int nt) {
  const v16u* p = (const v16u*)packed + ((size_t)(kt * 16 + nt) * 32 + (threadIdx.x & 31));
  return __builtin_bit_cast(v16bf, *p);
}

__device__ __forceinline__ v8f wmma_bf16(v16bf a, v16bf b, v8f c) {
  return __builtin_amdgcn_wmma_f32_16x16x32_bf16(false, a, false, b, (short)0, c, false, false);
}

// ---------------- precompute kernels ----------------

// pack f32 row-major [K x 256] weight into bf16 fragment-major B layout:
// out[(((kt*16)+nt)*32 + lane)*16 + i] = W[(kt*32 + (lane>>4)*8 + (i&7) + (i>>3)*16)*256 + nt*16 + (lane&15)]
__global__ __launch_bounds__(256) void pack_b_kernel(const float* __restrict__ W, int K,
                                                     unsigned short* __restrict__ out) {
  int idx = blockIdx.x * blockDim.x + threadIdx.x;
  if (idx >= K * 256) return;
  int i = idx & 15;
  int lane = (idx >> 4) & 31;
  int ntkt = idx >> 9;
  int nt = ntkt & 15, kt = ntkt >> 4;
  int col = nt * 16 + (lane & 15);
  int k = kt * 32 + ((lane >> 4) << 3) + (i & 7) + ((i >> 3) << 4);
  out[idx] = f2bf(W[(size_t)k * 256 + col]);
}

// second moments of x (Px3): stats[0..2]=sum x_k, stats[3..8]=sum x_k*x_l (00,01,02,11,12,22)
__global__ __launch_bounds__(256) void bn_stats_kernel(const float* __restrict__ x, int P,
                                                       float* __restrict__ stats) {
  float a[9];
#pragma unroll
  for (int i = 0; i < 9; ++i) a[i] = 0.f;
  for (long long i = (long long)blockIdx.x * blockDim.x + threadIdx.x; i < P;
       i += (long long)gridDim.x * blockDim.x) {
    float x0 = x[i * 3 + 0], x1 = x[i * 3 + 1], x2 = x[i * 3 + 2];
    a[0] += x0; a[1] += x1; a[2] += x2;
    a[3] += x0 * x0; a[4] += x0 * x1; a[5] += x0 * x2;
    a[6] += x1 * x1; a[7] += x1 * x2; a[8] += x2 * x2;
  }
  __shared__ float red[256];
  int tid = threadIdx.x;
  for (int q = 0; q < 9; ++q) {
    red[tid] = a[q];
    __syncthreads();
    for (int s = 128; s > 0; s >>= 1) {
      if (tid < s) red[tid] += red[tid + s];
      __syncthreads();
    }
    if (tid == 0) atomicAdd(&stats[q], red[0]);
    __syncthreads();
  }
}

// fold batchnorm (analytic mean/var from moments) into W1'(3x256), b1'(256)
__global__ void fold_bn_kernel(const float* __restrict__ stats, const float* __restrict__ W1,
                               const float* __restrict__ b1, const float* __restrict__ g,
                               const float* __restrict__ bb, float invP,
                               float* __restrict__ W1p, float* __restrict__ b1p) {
  int j = threadIdx.x;
  float m0 = stats[0] * invP, m1 = stats[1] * invP, m2 = stats[2] * invP;
  float c00 = stats[3] * invP - m0 * m0, c01 = stats[4] * invP - m0 * m1;
  float c02 = stats[5] * invP - m0 * m2, c11 = stats[6] * invP - m1 * m1;
  float c12 = stats[7] * invP - m1 * m2, c22 = stats[8] * invP - m2 * m2;
  float w0 = W1[j], w1 = W1[256 + j], w2 = W1[512 + j];
  float mj = b1[j] + m0 * w0 + m1 * w1 + m2 * w2;
  float var = w0 * w0 * c00 + w1 * w1 * c11 + w2 * w2 * c22 +
              2.f * (w0 * w1 * c01 + w0 * w2 * c02 + w1 * w2 * c12);
  float s = g[j] * rsqrtf(var + EPSV);
  W1p[j] = w0 * s; W1p[256 + j] = w1 * s; W1p[512 + j] = w2 * s;
  b1p[j] = (b1[j] - mj) * s + bb[j];
}

// C[MxN=256] = A[MxK] @ B[Kx256]   (small weight-fold GEMMs, f32)
__global__ __launch_bounds__(256) void fold_gemm_kernel(const float* __restrict__ A,
                                                        const float* __restrict__ B,
                                                        float* __restrict__ C, int M, int K) {
  int idx = blockIdx.x * blockDim.x + threadIdx.x;
  if (idx >= M * 256) return;
  int m = idx >> 8, n = idx & 255;
  float s = 0.f;
  for (int k = 0; k < K; ++k) s = fmaf(A[(size_t)m * K + k], B[(size_t)k * 256 + n], s);
  C[idx] = s;
}

__global__ void node_bias_kernel(const float* __restrict__ nf_b, const float* __restrict__ sem_b,
                                 const float* __restrict__ nfW, float* __restrict__ out) {
  int j = threadIdx.x;
  float s = nf_b[j];
  for (int k = 0; k < 256; ++k) s = fmaf(sem_b[k], nfW[(size_t)k * 256 + j], s);
  out[j] = s;
}

__global__ void edge_bias_kernel(const float* __restrict__ ef_b, const float* __restrict__ epb2,
                                 const float* __restrict__ eab, const float* __restrict__ epob,
                                 const float* __restrict__ efW, float* __restrict__ out) {
  int j = threadIdx.x;
  float s = ef_b[j];
  for (int k = 0; k < 256; ++k) {
    s = fmaf(epb2[k], efW[(size_t)(256 + k) * 256 + j], s);
    s = fmaf(eab[k],  efW[(size_t)(512 + k) * 256 + j], s);
    s = fmaf(epob[k], efW[(size_t)(768 + k) * 256 + j], s);
  }
  out[j] = s;
}

// quadratic-form constants for analytic per-row LN of v5@epW1+b1:
// q[0..24]=W1*W1^T (5x5), q[25..29]=W1@b1, q[30]=|b1|^2, q[31..35]=row sums of W1, q[36]=sum b1
__global__ void ep_stats_kernel(const float* __restrict__ W1, const float* __restrict__ b1,
                                float* __restrict__ q) {
  int t = threadIdx.x;
  if (t < 25) {
    int a = t / 5, b = t % 5; float s = 0.f;
    for (int j = 0; j < 256; ++j) s += W1[a * 256 + j] * W1[b * 256 + j];
    q[t] = s;
  } else if (t < 30) {
    int a = t - 25; float s = 0.f;
    for (int j = 0; j < 256; ++j) s += W1[a * 256 + j] * b1[j];
    q[t] = s;
  } else if (t == 30) {
    float s = 0.f;
    for (int j = 0; j < 256; ++j) s += b1[j] * b1[j];
    q[30] = s;
  } else if (t < 36) {
    int a = t - 31; float s = 0.f;
    for (int j = 0; j < 256; ++j) s += W1[a * 256 + j];
    q[t] = s;
  } else if (t == 36) {
    float s = 0.f;
    for (int j = 0; j < 256; ++j) s += b1[j];
    q[36] = s;
  }
}

// ---------------- fused point-cloud encoder ----------------
// per block: 64 points. h1'=relu(x@W1'+b1') (BN folded) -> bf16 LDS,
// h2 = relu(h1'@W2+b2) via WMMA, segment-max via int atomicMax (vals >= 0).
__global__ __launch_bounds__(128) void point_enc_kernel(
    const float* __restrict__ x, const int* __restrict__ bidx,
    const float* __restrict__ W1p, const float* __restrict__ b1p,
    const unsigned short* __restrict__ W2pk, const float* __restrict__ b2,
    float* __restrict__ nodebuf, int P) {
  __shared__ float xs[64][3];
  __shared__ int bix[64];
  __shared__ __align__(16) unsigned short h1[64][264];
  const int tid = threadIdx.x;
  const long long base = (long long)blockIdx.x * 64;
  if (tid < 64) {
    long long g = base + tid;
    bool v = g < P;
    xs[tid][0] = v ? x[g * 3 + 0] : 0.f;
    xs[tid][1] = v ? x[g * 3 + 1] : 0.f;
    xs[tid][2] = v ? x[g * 3 + 2] : 0.f;
    bix[tid] = v ? bidx[g] : 0;
  }
  __syncthreads();
  for (int e = tid; e < 64 * 128; e += 128) {
    int r = e >> 7, j = (e & 127) * 2;
    float x0 = xs[r][0], x1 = xs[r][1], x2 = xs[r][2];
    float h0 = fmaf(x0, W1p[j],     fmaf(x1, W1p[256 + j],     fmaf(x2, W1p[512 + j],     b1p[j])));
    float h1v = fmaf(x0, W1p[j + 1], fmaf(x1, W1p[257 + j],     fmaf(x2, W1p[513 + j],     b1p[j + 1])));
    *(unsigned int*)&h1[r][j] = pack2bf(fmaxf(h0, 0.f), fmaxf(h1v, 0.f));
  }
  __syncthreads();
  const int wave = tid >> 5, lane = tid & 31, half = lane >> 4, l15 = lane & 15;
  const unsigned short* Arow = &h1[wave * 16][0];
  v8f acc[16];
#pragma unroll
  for (int nt = 0; nt < 16; ++nt) acc[nt] = zero8();
  for (int k = 0; k < 8; ++k) {
    v16bf a = load_a_lds(Arow, 264, k * 32);
#pragma unroll
    for (int nt = 0; nt < 16; ++nt) {
      v16bf b = load_b_pk(W2pk, k, nt);
      acc[nt] = wmma_bf16(a, b, acc[nt]);
    }
  }
#pragma unroll
  for (int nt = 0; nt < 16; ++nt) {
    int col = nt * 16 + l15;
    float bb = b2[col];
#pragma unroll
    for (int i = 0; i < 8; ++i) {
      int r = wave * 16 + i + 8 * half;
      long long g = base + r;
      if (g < P) {
        float val = fmaxf(acc[nt][i] + bb, 0.f);  // relu >= 0 -> int compare valid
        atomicMax((int*)&nodebuf[(size_t)bix[r] * 256 + col], __float_as_int(val));
      }
    }
  }
}

// ---------------- fused node head ----------------
// acc = x_sem@(semW@nfA) + h_pos@nfB + h_aff@nfC  (K=1024), + bias, LN, relu
__global__ __launch_bounds__(128) void node_kernel(
    const float* __restrict__ xsem, const float* __restrict__ npos, const float* __restrict__ naff,
    const unsigned short* __restrict__ WsemPk, const unsigned short* __restrict__ nfBpk,
    const unsigned short* __restrict__ nfCpk, const float* __restrict__ nbias,
    const float* __restrict__ g, const float* __restrict__ lb, float* __restrict__ out, int N) {
  __shared__ __align__(16) unsigned short A[64][264];
  const int tid = threadIdx.x;
  const long long base = (long long)blockIdx.x * 64;
  const int wave = tid >> 5, lane = tid & 31, half = lane >> 4, l15 = lane & 15;
  v8f acc[16];
#pragma unroll
  for (int nt = 0; nt < 16; ++nt) acc[nt] = zero8();
  for (int c = 0; c < 4; ++c) {
    const float* src = (c <= 1) ? xsem : (c == 2 ? npos : naff);
    const int rstride = (c <= 1) ? 512 : 256;
    const int coff = (c == 1) ? 256 : 0;
    const unsigned short* B = (c <= 1) ? WsemPk : (c == 2 ? nfBpk : nfCpk);
    const int ktbase = (c == 1) ? 8 : 0;
    __syncthreads();
    for (int e = tid; e < 64 * 128; e += 128) {
      int r = e >> 7, j = (e & 127) * 2;
      long long gg = base + r;
      float v0 = 0.f, v1 = 0.f;
      if (gg < N) {
        const float* s = src + (size_t)gg * rstride + coff + j;
        v0 = s[0]; v1 = s[1];
      }
      *(unsigned int*)&A[r][j] = pack2bf(v0, v1);
    }
    __syncthreads();
    for (int k = 0; k < 8; ++k) {
      v16bf a = load_a_lds(&A[wave * 16][0], 264, k * 32);
#pragma unroll
      for (int nt = 0; nt < 16; ++nt) {
        v16bf b = load_b_pk(B, ktbase + k, nt);
        acc[nt] = wmma_bf16(a, b, acc[nt]);
      }
    }
  }
#pragma unroll
  for (int nt = 0; nt < 16; ++nt) {
    float bb = nbias[nt * 16 + l15];
#pragma unroll
    for (int i = 0; i < 8; ++i) acc[nt][i] += bb;
  }
  for (int i = 0; i < 8; ++i) {  // in-register LN across the 16-lane half
    float s = 0.f, q = 0.f;
#pragma unroll
    for (int nt = 0; nt < 16; ++nt) { float v = acc[nt][i]; s += v; q += v * v; }
    for (int m = 1; m < 16; m <<= 1) { s += __shfl_xor(s, m); q += __shfl_xor(q, m); }
    float mean = s * (1.f / 256.f);
    float var = q * (1.f / 256.f) - mean * mean;
    float inv = rsqrtf(var + EPSV);
    long long grow = base + wave * 16 + i + 8 * half;
    if (grow < N) {
      float* orow = out + (size_t)grow * 256;
#pragma unroll
      for (int nt = 0; nt < 16; ++nt) {
        int col = nt * 16 + l15;
        float v = (acc[nt][i] - mean) * inv * g[col] + lb[col];
        orow[col] = fmaxf(v, 0.f);
      }
    }
  }
}

// ---------------- fused edge head ----------------
__global__ __launch_bounds__(128) void edge_kernel(
    const float* __restrict__ eparam, const int* __restrict__ etype,
    const float* __restrict__ anchor, const float* __restrict__ pose,
    const float* __restrict__ epW1, const float* __restrict__ epb1,
    const float* __restrict__ epg, const float* __restrict__ epb,
    const float* __restrict__ epq, const unsigned short* __restrict__ W2pk,
    const float* __restrict__ ttab, const float* __restrict__ WAf,
    const float* __restrict__ WPf, const float* __restrict__ ebias,
    const float* __restrict__ efg, const float* __restrict__ efb,
    float* __restrict__ outE, int E) {
  __shared__ float v5[64][5];
  __shared__ float anc[64][24];
  __shared__ float pse[64][9];
  __shared__ int ety[64];
  __shared__ float rm[64], ri[64];
  __shared__ __align__(16) unsigned short A[64][264];
  const int tid = threadIdx.x;
  const long long base = (long long)blockIdx.x * 64;
  if (tid < 64) {
    long long g = base + tid;
    bool v = g < E;
    float d = v ? eparam[g * 3 + 0] : 0.f;
    float th = v ? eparam[g * 3 + 1] : 0.f;
    float ph = v ? eparam[g * 3 + 2] : 0.f;
    v5[tid][0] = fminf(fmaxf(d * 10.f, -5.f), 5.f);
    v5[tid][1] = sinf(th); v5[tid][2] = cosf(th);
    v5[tid][3] = sinf(ph); v5[tid][4] = cosf(ph);
    ety[tid] = v ? etype[g] : 0;
  }
  for (int e = tid; e < 64 * 24; e += 128) {
    int r = e / 24, k = e % 24;
    long long g = base + r;
    anc[r][k] = (g < E) ? anchor[g * 24 + k] : 0.f;
  }
  for (int e = tid; e < 64 * 9; e += 128) {
    int r = e / 9, k = e % 9;
    long long g = base + r;
    pse[r][k] = (g < E) ? pose[g * 9 + k] : 0.f;
  }
  __syncthreads();
  if (tid < 64) {  // analytic per-row LN stats from quadratic form
    float rs = epq[36], qq = epq[30];
#pragma unroll
    for (int a = 0; a < 5; ++a) {
      float va = v5[tid][a];
      rs += va * epq[31 + a];
      qq += 2.f * va * epq[25 + a];
#pragma unroll
      for (int b = 0; b < 5; ++b) qq += va * v5[tid][b] * epq[a * 5 + b];
    }
    float mean = rs * (1.f / 256.f);
    float var = qq * (1.f / 256.f) - mean * mean;
    rm[tid] = mean;
    ri[tid] = rsqrtf(var + EPSV);
  }
  __syncthreads();
  for (int e = tid; e < 64 * 128; e += 128) {
    int r = e >> 7, j = (e & 127) * 2;
    float h0 = epb1[j], h1v = epb1[j + 1];
#pragma unroll
    for (int k = 0; k < 5; ++k) {
      float vk = v5[r][k];
      h0 = fmaf(vk, epW1[k * 256 + j], h0);
      h1v = fmaf(vk, epW1[k * 256 + j + 1], h1v);
    }
    float m = rm[r], inv = ri[r];
    float a0 = fmaxf((h0 - m) * inv * epg[j] + epb[j], 0.f);
    float a1 = fmaxf((h1v - m) * inv * epg[j + 1] + epb[j + 1], 0.f);
    *(unsigned int*)&A[r][j] = pack2bf(a0, a1);
  }
  __syncthreads();
  const int wave = tid >> 5, lane = tid & 31, half = lane >> 4, l15 = lane & 15;
  v8f acc[16];
#pragma unroll
  for (int nt = 0; nt < 16; ++nt) acc[nt] = zero8();
  for (int k = 0; k < 8; ++k) {
    v16bf a = load_a_lds(&A[wave * 16][0], 264, k * 32);
#pragma unroll
    for (int nt = 0; nt < 16; ++nt) {
      v16bf b = load_b_pk(W2pk, k, nt);
      acc[nt] = wmma_bf16(a, b, acc[nt]);
    }
  }
  // add folded type-table + anchor/pose contributions + bias (K=24/K=9 VALU dots)
  for (int nt = 0; nt < 16; ++nt) {
    int col = nt * 16 + l15;
    float wa[24], wp[9];
#pragma unroll
    for (int k = 0; k < 24; ++k) wa[k] = WAf[(size_t)k * 256 + col];
#pragma unroll
    for (int k = 0; k < 9; ++k) wp[k] = WPf[(size_t)k * 256 + col];
    float eb = ebias[col];
#pragma unroll
    for (int i = 0; i < 8; ++i) {
      int r = wave * 16 + i + 8 * half;
      float ex = eb + ttab[(size_t)ety[r] * 256 + col];
#pragma unroll
      for (int k = 0; k < 24; ++k) ex = fmaf(anc[r][k], wa[k], ex);
#pragma unroll
      for (int k = 0; k < 9; ++k) ex = fmaf(pse[r][k], wp[k], ex);
      acc[nt][i] += ex;
    }
  }
  for (int i = 0; i < 8; ++i) {
    float s = 0.f, q = 0.f;
#pragma unroll
    for (int nt = 0; nt < 16; ++nt) { float v = acc[nt][i]; s += v; q += v * v; }
    for (int m = 1; m < 16; m <<= 1) { s += __shfl_xor(s, m); q += __shfl_xor(q, m); }
    float mean = s * (1.f / 256.f);
    float var = q * (1.f / 256.f) - mean * mean;
    float inv = rsqrtf(var + EPSV);
    long long g = base + wave * 16 + i + 8 * half;
    if (g < E) {
      float* orow = outE + (size_t)g * 256;
#pragma unroll
      for (int nt = 0; nt < 16; ++nt) {
        int col = nt * 16 + l15;
        float v = (acc[nt][i] - mean) * inv * efg[col] + efb[col];
        orow[col] = fmaxf(v, 0.f);
      }
    }
  }
}

extern "C" void kernel_launch(void* const* d_in, const int* in_sizes, int n_in,
                              void* d_out, int out_size, void* d_ws, size_t ws_size,
                              hipStream_t stream) {
  (void)n_in; (void)out_size; (void)ws_size;
  const float* x_sem   = (const float*)d_in[0];
  const float* x_pos   = (const float*)d_in[1];
  const int*   pos_idx = (const int*)d_in[2];
  const float* x_aff   = (const float*)d_in[3];
  const int*   aff_idx = (const int*)d_in[4];
  const int*   etype   = (const int*)d_in[5];
  const float* eparam  = (const float*)d_in[6];
  const float* eanchor = (const float*)d_in[7];
  const float* epose   = (const float*)d_in[8];
  const float* sem_W   = (const float*)d_in[10];
  const float* sem_b   = (const float*)d_in[11];
  const float* pos_W1  = (const float*)d_in[12];
  const float* pos_b1  = (const float*)d_in[13];
  const float* pos_bng = (const float*)d_in[14];
  const float* pos_bnb = (const float*)d_in[15];
  const float* pos_W2  = (const float*)d_in[16];
  const float* pos_b2  = (const float*)d_in[17];
  const float* aff_W1  = (const float*)d_in[18];
  const float* aff_b1  = (const float*)d_in[19];
  const float* aff_bng = (const float*)d_in[20];
  const float* aff_bnb = (const float*)d_in[21];
  const float* aff_W2  = (const float*)d_in[22];
  const float* aff_b2  = (const float*)d_in[23];
  const float* nf_W    = (const float*)d_in[24];
  const float* nf_b    = (const float*)d_in[25];
  const float* nf_lng  = (const float*)d_in[26];
  const float* nf_lnb  = (const float*)d_in[27];
  const float* et_emb  = (const float*)d_in[28];
  const float* ep_W1   = (const float*)d_in[29];
  const float* ep_b1   = (const float*)d_in[30];
  const float* ep_lng  = (const float*)d_in[31];
  const float* ep_lnb  = (const float*)d_in[32];
  const float* ep_W2   = (const float*)d_in[33];
  const float* ep_b2   = (const float*)d_in[34];
  const float* ea_W    = (const float*)d_in[35];
  const float* ea_b    = (const float*)d_in[36];
  const float* epo_W   = (const float*)d_in[37];
  const float* epo_b   = (const float*)d_in[38];
  const float* ef_W    = (const float*)d_in[39];
  const float* ef_b    = (const float*)d_in[40];
  const float* ef_lng  = (const float*)d_in[41];
  const float* ef_lnb  = (const float*)d_in[42];

  const int N = in_sizes[0] / 512;
  const int P = in_sizes[1] / 3;
  const int E = in_sizes[5];

  // ---- workspace layout (floats; every block multiple of 8 floats = 32B aligned) ----
  float* wsf = (float*)d_ws;
  size_t o = 0;
  float* node_pos = wsf + o; o += (size_t)N * 256;
  float* node_aff = wsf + o; o += (size_t)N * 256;
  float* stats    = wsf + o; o += 32;   // pos @0..8, aff @16..24
  float* W1p_pos  = wsf + o; o += 768;
  float* b1p_pos  = wsf + o; o += 256;
  float* W1p_aff  = wsf + o; o += 768;
  float* b1p_aff  = wsf + o; o += 256;
  float* ttab     = wsf + o; o += 7 * 256;
  float* WAf      = wsf + o; o += 24 * 256;
  float* WPf      = wsf + o; o += 9 * 256;
  float* nbias    = wsf + o; o += 256;
  float* ebias    = wsf + o; o += 256;
  float* epq      = wsf + o; o += 40;
  float* tmpf     = wsf + o; o += 512 * 256;
  unsigned short* wsb = (unsigned short*)(wsf + o);   // 32B aligned
  size_t ob = 0;
  unsigned short* W2pos_pk = wsb + ob; ob += 256 * 256;
  unsigned short* W2aff_pk = wsb + ob; ob += 256 * 256;
  unsigned short* Wsem_pk  = wsb + ob; ob += 512 * 256;
  unsigned short* nfB_pk   = wsb + ob; ob += 256 * 256;
  unsigned short* nfC_pk   = wsb + ob; ob += 256 * 256;
  unsigned short* epW2_pk  = wsb + ob; ob += 256 * 256;

  // zero segment-max accumulators + stats (relu outputs >= 0 -> 0-init == -inf semantics)
  hipMemsetAsync(d_ws, 0, ((size_t)2 * N * 256 + 64) * sizeof(float), stream);

  // ---- precompute: BN moments + folds + fragment-packed bf16 weights ----
  bn_stats_kernel<<<256, 256, 0, stream>>>(x_pos, P, stats);
  bn_stats_kernel<<<256, 256, 0, stream>>>(x_aff, P, stats + 16);
  fold_bn_kernel<<<1, 256, 0, stream>>>(stats, pos_W1, pos_b1, pos_bng, pos_bnb,
                                        1.f / (float)P, W1p_pos, b1p_pos);
  fold_bn_kernel<<<1, 256, 0, stream>>>(stats + 16, aff_W1, aff_b1, aff_bng, aff_bnb,
                                        1.f / (float)P, W1p_aff, b1p_aff);

  fold_gemm_kernel<<<512, 256, 0, stream>>>(sem_W, nf_W, tmpf, 512, 256);   // semW @ nfW_A
  pack_b_kernel<<<512, 256, 0, stream>>>(tmpf, 512, Wsem_pk);
  fold_gemm_kernel<<<256, 256, 0, stream>>>(ep_W2, ef_W + (size_t)256 * 256, tmpf, 256, 256);
  pack_b_kernel<<<256, 256, 0, stream>>>(tmpf, 256, epW2_pk);
  fold_gemm_kernel<<<24, 256, 0, stream>>>(ea_W, ef_W + (size_t)512 * 256, WAf, 24, 256);
  fold_gemm_kernel<<<9, 256, 0, stream>>>(epo_W, ef_W + (size_t)768 * 256, WPf, 9, 256);
  fold_gemm_kernel<<<7, 256, 0, stream>>>(et_emb, ef_W, ttab, 7, 256);

  node_bias_kernel<<<1, 256, 0, stream>>>(nf_b, sem_b, nf_W, nbias);
  edge_bias_kernel<<<1, 256, 0, stream>>>(ef_b, ep_b2, ea_b, epo_b, ef_W, ebias);
  ep_stats_kernel<<<1, 64, 0, stream>>>(ep_W1, ep_b1, epq);

  pack_b_kernel<<<256, 256, 0, stream>>>(pos_W2, 256, W2pos_pk);
  pack_b_kernel<<<256, 256, 0, stream>>>(aff_W2, 256, W2aff_pk);
  pack_b_kernel<<<256, 256, 0, stream>>>(nf_W + (size_t)256 * 256, 256, nfB_pk);
  pack_b_kernel<<<256, 256, 0, stream>>>(nf_W + (size_t)512 * 256, 256, nfC_pk);

  // ---- heavy fused kernels ----
  point_enc_kernel<<<(P + 63) / 64, 128, 0, stream>>>(x_pos, pos_idx, W1p_pos, b1p_pos,
                                                      W2pos_pk, pos_b2, node_pos, P);
  point_enc_kernel<<<(P + 63) / 64, 128, 0, stream>>>(x_aff, aff_idx, W1p_aff, b1p_aff,
                                                      W2aff_pk, aff_b2, node_aff, P);
  node_kernel<<<(N + 63) / 64, 128, 0, stream>>>(x_sem, node_pos, node_aff, Wsem_pk, nfB_pk,
                                                 nfC_pk, nbias, nf_lng, nf_lnb,
                                                 (float*)d_out, N);
  edge_kernel<<<(E + 63) / 64, 128, 0, stream>>>(eparam, etype, eanchor, epose, ep_W1, ep_b1,
                                                 ep_lng, ep_lnb, epq, epW2_pk, ttab, WAf, WPf,
                                                 ebias, ef_lng, ef_lnb,
                                                 (float*)d_out + (size_t)N * 256, E);
}
